// TheoSCAM_45930380264377
// MI455X (gfx1250) — matrix operand order repaired and checked by reading, not compiled, not run on hardware
//
#include <hip/hip_runtime.h>

typedef __attribute__((ext_vector_type(16))) _Float16 v16h;
typedef __attribute__((ext_vector_type(8)))  _Float16 v8h;
typedef __attribute__((ext_vector_type(8)))  float    v8f;
typedef __attribute__((ext_vector_type(4)))  float    v4f;

#define B_Q   2048
#define N_K   65536
#define D_DIM 128
#define NS    32                           // N splits
#define NKEYS_PER_SPLIT (N_K / NS)         // 2048 keys per split
#define CHUNK 64                           // keys staged in LDS per step
#define NCHUNK (NKEYS_PER_SPLIT / CHUNK)   // 32
#define LDS_STRIDE 136                     // 128 halfs + 8 pad (272B: rotates banks)
#define MQ_PER_BLOCK 128                   // 8 waves x 16 queries

#define KEY_CVT_BLOCKS ((N_K * D_DIM) / (256 * 8))   // 4096
#define BIAS_BLOCKS    (N_K / (256 * 4))             // 64

// ---------------------------------------------------------------------------
// CDNA5 async-copy (global -> LDS, bypasses VGPRs, tracked by ASYNCcnt).
// Per-lane LDS destination address allows scatter into the padded layout.
// ---------------------------------------------------------------------------
__device__ __forceinline__ void async_to_lds_b128(unsigned lds_byte_off,
                                                  unsigned gvoff,
                                                  const void* gbase)
{
    asm volatile("global_load_async_to_lds_b128 %0, %1, %2"
                 :: "v"(lds_byte_off), "v"(gvoff), "s"(gbase)
                 : "memory");
}
__device__ __forceinline__ void wait_asynccnt0()
{
    asm volatile("s_wait_asynccnt 0" ::: "memory");
}

// ---------------------------------------------------------------------------
// Kernel 0 (prep): keys f32 -> f16 into ws, and is_active -> f32 bias
// (0.0 if active, -inf if not). Both become async-copyable, L2-resident.
// ---------------------------------------------------------------------------
__global__ __launch_bounds__(256)
void TheoSCAM_prep(const float* __restrict__ keys,
                   const unsigned char* __restrict__ active,
                   _Float16* __restrict__ keys16,
                   float* __restrict__ bias32)
{
    if (blockIdx.x < KEY_CVT_BLOCKS) {
        size_t base = ((size_t)blockIdx.x * 256 + threadIdx.x) * 8;
        v4f f0 = *(const v4f*)(keys + base);
        v4f f1 = *(const v4f*)(keys + base + 4);
        v8h h;
#pragma unroll
        for (int i = 0; i < 4; ++i) {
            h[i]     = (_Float16)f0[i];
            h[4 + i] = (_Float16)f1[i];
        }
        *(v8h*)(keys16 + base) = h;
    } else {
        int i = (blockIdx.x - KEY_CVT_BLOCKS) * 256 + threadIdx.x; // 4 bools each
        unsigned a4 = *(const unsigned*)(active + (size_t)i * 4);
        v4f b;
#pragma unroll
        for (int j = 0; j < 4; ++j)
            b[j] = ((a4 >> (8 * j)) & 0xffu) ? 0.0f : -__builtin_inff();
        *(v4f*)(bias32 + (size_t)i * 4) = b;
    }
}

// ---------------------------------------------------------------------------
// Kernel 1: WMMA f16 scoring (sim^T tiles) fused with masked running argmax.
// Grid: (B/128, NS). Block: 256 threads = 8 wave32, each wave owns 16 queries.
// A = key tile (16x32 f16, async-staged LDS), B = query fragment (resident).
// Activity bias (0 / -inf) is async-staged too and used as the WMMA C
// initializer, so masking costs zero VALU ops (matches masked_fill(-inf)).
// ---------------------------------------------------------------------------
__global__ __launch_bounds__(256)
void TheoSCAM_score_argmax(const float* __restrict__ spikes,
                           const _Float16* __restrict__ keys16,
                           const float* __restrict__ bias32,
                           float* __restrict__ pmax,
                           int*   __restrict__ pidx)
{
    __shared__ _Float16 skey[2][CHUNK * LDS_STRIDE];   // 2 x 17408 B
    __shared__ float    sbias[2][CHUNK];               // 0.0f or -inf per key

    const int tid  = threadIdx.x;
    const int lane = tid & 31;
    const int wave = tid >> 5;
    const int mb   = blockIdx.x;            // 0..15
    const int ns   = blockIdx.y;            // 0..31
    const int keyBase = ns * NKEYS_PER_SPLIT;

    // ---- resident B fragments (queries), f32 -> f16 once ----
    const int q     = mb * MQ_PER_BLOCK + wave * 16 + (lane & 15);
    const int bBase = (lane < 16) ? 0 : 16;
    v16h bfrag[4];
#pragma unroll
    for (int s = 0; s < 4; ++s) {
        const float* src = spikes + (size_t)q * D_DIM + s * 32 + bBase;
        v4f f0 = *(const v4f*)(src + 0);
        v4f f1 = *(const v4f*)(src + 4);
        v4f f2 = *(const v4f*)(src + 8);
        v4f f3 = *(const v4f*)(src + 12);
        v16h b;
#pragma unroll
        for (int i = 0; i < 4; ++i) {
            b[i]      = (_Float16)f0[i];
            b[4 + i]  = (_Float16)f1[i];
            b[8 + i]  = (_Float16)f2[i];
            b[12 + i] = (_Float16)f3[i];
        }
        bfrag[s] = b;
    }

    float best   = -__builtin_inff();
    int   bestid = keyBase;                 // argmax of all -inf -> first key

    const unsigned keyLds[2] = {
        (unsigned)(size_t)&skey[0][0], (unsigned)(size_t)&skey[1][0]
    };
    const unsigned biasLds[2] = {
        (unsigned)(size_t)&sbias[0][0], (unsigned)(size_t)&sbias[1][0]
    };

    // ---- stage one 64-key chunk: fully asynchronous (keys + bias) ----
    auto stage = [&](int chunk, int buf) {
        const int rowStartKey = keyBase + chunk * CHUNK;
        // keys: CHUNK*256B = 1024 x 16B transfers, 256 threads -> 4 each
#pragma unroll
        for (int it = 0; it < 4; ++it) {
            int u   = it * 256 + tid;       // 16B-unit index
            int row = u >> 4;               // 16 x 16B per key row
            int c16 = u & 15;
            unsigned gvoff = (unsigned)((rowStartKey + row) * 256 + c16 * 16);
            unsigned loff  = keyLds[buf] + (unsigned)(row * (LDS_STRIDE * 2) + c16 * 16);
            async_to_lds_b128(loff, gvoff, (const void*)keys16);
        }
        // bias: CHUNK*4B = 256B = 16 x 16B transfers (threads 0..15)
        if (tid < CHUNK / 4) {
            unsigned gvoff = (unsigned)(rowStartKey * 4 + tid * 16);
            unsigned loff  = biasLds[buf] + (unsigned)(tid * 16);
            async_to_lds_b128(loff, gvoff, (const void*)bias32);
        }
    };

    // ---- consume one chunk: 4 key tiles x (4 WMMA + max-tree merge) ----
    auto compute = [&](int chunk, int buf) {
        const int aBase   = (lane < 16) ? 0 : 16;  // A 16x32 contiguous half-K pattern
        const int halfOff = (lane < 16) ? 0 : 8;   // C: lanes>=16 hold M=j+8
        const int m       = lane & 15;
#pragma unroll
        for (int t = 0; t < 4; ++t) {
            const _Float16* rowp = &skey[buf][(t * 16 + m) * LDS_STRIDE];
            // preload C bias + all A fragments, then run WMMAs back-to-back
            v4f b0 = *(const v4f*)(&sbias[buf][t * 16 + halfOff]);
            v4f b1 = *(const v4f*)(&sbias[buf][t * 16 + halfOff + 4]);
            v8h alo[4], ahi[4];
#pragma unroll
            for (int s = 0; s < 4; ++s) {
                alo[s] = *(const v8h*)(rowp + s * 32 + aBase);       // ds_load_b128
                ahi[s] = *(const v8h*)(rowp + s * 32 + aBase + 8);   // ds_load_b128
            }
            // C initialized with activity bias: 0.0 (active) or -inf (inactive)
            v8f c = __builtin_shufflevector(b0, b1, 0, 1, 2, 3, 4, 5, 6, 7);
#pragma unroll
            for (int s = 0; s < 4; ++s) {
                v16h a = __builtin_shufflevector(alo[s], ahi[s],
                        0, 1, 2, 3, 4, 5, 6, 7, 8, 9, 10, 11, 12, 13, 14, 15);
                c = __builtin_amdgcn_wmma_f32_16x16x32_f16(
                        false, a, false, bfrag[s], (short)0, c, false, false);
            }
            // tile max via tree (short dep chain), index scan only on improve
            float m0 = fmaxf(c[0], c[1]);
            float m1 = fmaxf(c[2], c[3]);
            float m2 = fmaxf(c[4], c[5]);
            float m3 = fmaxf(c[6], c[7]);
            float tmax = fmaxf(fmaxf(m0, m1), fmaxf(m2, m3));
            if (tmax > best) {              // execz-skippable rare path
                const int tileKey = keyBase + chunk * CHUNK + t * 16 + halfOff;
#pragma unroll
                for (int j = 0; j < 8; ++j) {
                    float sc = c[j];
                    if (sc > best) { best = sc; bestid = tileKey + j; }
                }
            }
        }
    };

    // ---- double-buffered pipeline: wait own ASYNCcnt, then barrier ----
    stage(0, 0);
    for (int ch = 0; ch < NCHUNK; ++ch) {
        wait_asynccnt0();
        __syncthreads();
        if (ch + 1 < NCHUNK) stage(ch + 1, (ch + 1) & 1);
        compute(ch, ch & 1);
    }

    // ---- combine lane halves (keys j vs j+8 live in lanes L and L^16) ----
    float ob = __shfl_xor(best, 16, 32);
    int   oi = __shfl_xor(bestid, 16, 32);
    if (ob > best || (ob == best && oi < bestid)) { best = ob; bestid = oi; }

    if (lane < 16) {                       // lanes 0..15 hold the 16 queries
        pmax[(size_t)q * NS + ns] = best;
        pidx[(size_t)q * NS + ns] = bestid;
    }
}

// ---------------------------------------------------------------------------
// Kernel 2: reduce the NS=32 split partials per query (one wave32 per query),
// then gather action_values[best] (float4 per lane) and write outputs.
// d_out layout: retrieved[B*D] | confidence[B] | best_idx[B] (int bits).
// ---------------------------------------------------------------------------
__global__ __launch_bounds__(32)
void TheoSCAM_reduce_gather(const float* __restrict__ pmax,
                            const int*   __restrict__ pidx,
                            const float* __restrict__ avals,
                            float* __restrict__ out)
{
    const int q = blockIdx.x;
    const int t = threadIdx.x;             // 0..31 == NS
    float m  = pmax[(size_t)q * NS + t];
    int   id = pidx[(size_t)q * NS + t];
#pragma unroll
    for (int off = 16; off > 0; off >>= 1) {
        float om = __shfl_down(m, (unsigned)off, 32);
        int   oi = __shfl_down(id, (unsigned)off, 32);
        if (om > m || (om == m && oi < id)) { m = om; id = oi; }
    }
    id = __shfl(id, 0, 32);
    m  = __shfl(m, 0, 32);

    v4f v = *(const v4f*)(avals + (size_t)id * D_DIM + t * 4);
    *(v4f*)(out + (size_t)q * D_DIM + t * 4) = v;
    if (t == 0) {
        out[(size_t)B_Q * D_DIM + q] = m;
        ((int*)out)[(size_t)B_Q * D_DIM + B_Q + q] = id;
    }
}

extern "C" void kernel_launch(void* const* d_in, const int* in_sizes, int n_in,
                              void* d_out, int out_size, void* d_ws, size_t ws_size,
                              hipStream_t stream)
{
    const float*         spikes = (const float*)d_in[0];
    const float*         keys   = (const float*)d_in[1];
    const float*         avals  = (const float*)d_in[2];
    const unsigned char* active = (const unsigned char*)d_in[3];

    // ws layout: pmax (256KB) | pidx (256KB) | bias (256KB) | keys f16 (16MB)
    char* ws = (char*)d_ws;
    float*    pmax   = (float*)ws;
    int*      pidx   = (int*)(ws + 1u * 256 * 1024);
    float*    bias32 = (float*)(ws + 2u * 256 * 1024);
    _Float16* keys16 = (_Float16*)(ws + 3u * 256 * 1024);

    TheoSCAM_prep<<<KEY_CVT_BLOCKS + BIAS_BLOCKS, 256, 0, stream>>>(
        keys, active, keys16, bias32);

    dim3 grid1(B_Q / MQ_PER_BLOCK, NS);    // 16 x 32 = 512 workgroups
    TheoSCAM_score_argmax<<<grid1, 256, 0, stream>>>(spikes, keys16, bias32, pmax, pidx);
    TheoSCAM_reduce_gather<<<B_Q, 32, 0, stream>>>(pmax, pidx, avals, (float*)d_out);
}
